// DGMatch_65360812310794
// MI455X (gfx1250) — compile-verified, matrix operand order checked by reading, not run on hardware
//
#include <hip/hip_runtime.h>
#include <hip/hip_bf16.h>

// ---------------------------------------------------------------------------
// DGMatch forward on MI455X (gfx1250, wave32).
// Heavy GEMMs (edge MLPs, feat MLP, kNN Gram) use v_wmma_f32_16x16x32_f16.
// BN folded into f16 weights at prep; weights pre-swizzled into WMMA B-frag
// order (one contiguous 32B chunk per lane per 16x16x32 tile).
// Edge conv / feat MLP batch 4 M-tiles per workgroup so each weight fragment
// feeds 4 WMMAs (4x arithmetic intensity vs L2).
// ---------------------------------------------------------------------------

typedef _Float16 half_t;
typedef _Float16 __attribute__((ext_vector_type(16))) v16h;
typedef _Float16 __attribute__((ext_vector_type(8)))  v8h;
typedef float    __attribute__((ext_vector_type(8)))  v8f;

#define B_    8
#define N_    2048
#define NODES (B_ * N_)
#define E_    128
#define D2E   256
#define D4E   512
#define KNN   16
#define C_    21
#define NPW   4      // M-tiles (nodes / node-tiles) batched per workgroup
#define BN_INV 0.99999500003749972f  // 1/sqrt(1+1e-5)

static __device__ __forceinline__ v16h concat8(v8h lo, v8h hi) {
  union { v16h v; v8h h[2]; } u;
  u.h[0] = lo; u.h[1] = hi;
  return u.v;
}

// A-fragment (16x32 f16, row-major source with `pitch` halves per row).
// Lanes 0-15 -> M=lane, K={0..7,16..23}; lanes 16-31 -> K={8..15,24..31}.
static __device__ __forceinline__ v16h load_afrag(const half_t* base, int pitch,
                                                  int lane, int kt) {
  const int m  = lane & 15;
  const int k0 = (kt << 5) + ((lane & 16) ? 8 : 0);
  const half_t* p = base + m * pitch + k0;
  v8h lo = *(const v8h*)(p);
  v8h hi = *(const v8h*)(p + 16);
  return concat8(lo, hi);
}

// B-fragment from pre-swizzled weights: [(kt*NT+nt)*32 + lane] * 16 halves.
static __device__ __forceinline__ v16h load_bfrag(const half_t* wfr, int NT,
                                                  int kt, int nt, int lane) {
  return *(const v16h*)(wfr + (size_t)((((kt * NT + nt) << 5) + lane) << 4));
}

static __device__ __forceinline__ v8f wmma16(v16h a, v16h b, v8f c) {
  return __builtin_amdgcn_wmma_f32_16x16x32_f16(false, a, false, b,
                                                (short)0, c, false, false);
}

// ---------------------------------------------------------------------------
// Prep: fold BN into weights, convert to f16, swizzle into fragment order.
// ---------------------------------------------------------------------------
__global__ void prep_weights(const float* __restrict__ W, const float* __restrict__ bias,
                             const float* __restrict__ g, const float* __restrict__ beta,
                             int Ktot, int Ntot,
                             half_t* __restrict__ wfr, float* __restrict__ bout) {
  int i = blockIdx.x * blockDim.x + threadIdx.x;
  int total = Ktot * Ntot;
  if (i < total) {
    int k = i / Ntot, n = i - k * Ntot;
    float s = g ? g[n] * BN_INV : 1.0f;
    int kt = k >> 5, kin = k & 31, nt = n >> 4, nin = n & 15;
    int lane = nin + ((kin & 8) ? 16 : 0);
    int h = (kin & 7) | (((kin >> 4) & 1) << 3);
    size_t slot = ((size_t)(((kt * (Ntot >> 4) + nt) << 5) + lane) << 4) + h;
    wfr[slot] = (half_t)(W[i] * s);
  }
  if (i < Ntot) {
    float s = g ? g[i] * BN_INV : 1.0f;
    bout[i] = bias[i] * s + (beta ? beta[i] : 0.0f);
  }
}

// ---------------------------------------------------------------------------
// Encode: obj = class_table[ci] + pos_mlp(pos); feats = [obj, desc] (f16) + |x|^2
// ---------------------------------------------------------------------------
__global__ void encode_kernel(const int* __restrict__ ci, const float* __restrict__ pos,
                              const float* __restrict__ desc, const float* __restrict__ ctab,
                              const float* __restrict__ W0, const float* __restrict__ b0,
                              const float* __restrict__ W1, const float* __restrict__ b1,
                              half_t* __restrict__ feats, float* __restrict__ sq) {
  int n = blockIdx.x * blockDim.x + threadIdx.x;
  if (n >= NODES) return;
  int b = n / N_;
  float p0 = pos[n * 2 + 0], p1 = pos[n * 2 + 1];
  float h[64];
#pragma unroll
  for (int o = 0; o < 64; ++o) {
    float v = p0 * W0[o] + p1 * W0[64 + o] + b0[o];
    h[o] = v > 0.f ? v : 0.f;
  }
  int cls = ci[n];
  float ss = 0.f;
  for (int o = 0; o < 128; ++o) {
    float v = b1[o];
#pragma unroll
    for (int j = 0; j < 64; ++j) v += h[j] * W1[j * 128 + o];
    v = v > 0.f ? v : 0.f;
    v += ctab[cls * 128 + o];
    feats[(size_t)n * D2E + o] = (half_t)v;
    ss += v * v;
  }
  for (int o = 0; o < 128; ++o) {
    float v = desc[b * 128 + o];
    feats[(size_t)n * D2E + 128 + o] = (half_t)v;
    ss += v * v;
  }
  sq[n] = ss;
}

// ---------------------------------------------------------------------------
// kNN: per 16-query tile, Gram via WMMA (f16 in, f32 acc), d2 in LDS,
// serial insertion top-16 per row. Self forced to -inf (self-loop included).
// Block = 512 threads (16 waves); grid = B * N/16.
// ---------------------------------------------------------------------------
__global__ void knn_kernel(const half_t* __restrict__ X, const float* __restrict__ sq,
                           int* __restrict__ idxout) {
  __shared__ half_t Q[16 * D2E];      // 8 KB
  __shared__ float  D2[16 * 1024];    // 64 KB (column chunk of 1024)
  __shared__ float  SQQ[16];
  int blk = blockIdx.x;
  int b = blk >> 7;                   // / (N/16 = 128)
  int qt = blk & 127;
  int row0 = qt * 16;
  const half_t* Xb = X + (size_t)b * N_ * D2E;
  int tid = threadIdx.x, lane = tid & 31, wave = tid >> 5;

  {
    int e0 = tid * 8;                 // 512 threads x 8 halves = 16x256 tile
    int r = e0 >> 8, c = e0 & 255;
    *(v8h*)&Q[e0] = *(const v8h*)&Xb[(size_t)(row0 + r) * D2E + c];
  }
  if (tid < 16) SQQ[tid] = sq[b * N_ + row0 + tid];

  float bestd[KNN];
  int   besti[KNN];
  if (tid < 16) {
#pragma unroll
    for (int j = 0; j < KNN; ++j) { bestd[j] = 3.0e38f; besti[j] = 0; }
  }
  __syncthreads();

  for (int ch = 0; ch < 2; ++ch) {
    for (int t4 = 0; t4 < 4; ++t4) {
      int ct = ch * 64 + wave * 4 + t4;          // global column tile 0..127
      int col0 = ct * 16;
      v8f acc = {0.f, 0.f, 0.f, 0.f, 0.f, 0.f, 0.f, 0.f};
#pragma unroll
      for (int kt = 0; kt < 8; ++kt) {
        v16h a  = load_afrag(Q, D2E, lane, kt);
        v16h bf = load_afrag(Xb + (size_t)col0 * D2E, D2E, lane, kt);
        acc = wmma16(a, bf, acc);
      }
      int ncol = col0 + (lane & 15);
      float sqc = sq[b * N_ + ncol];
      int mb = (lane & 16) ? 8 : 0;
      int lcol = (ct - ch * 64) * 16 + (lane & 15);
#pragma unroll
      for (int i = 0; i < 8; ++i) {
        int m = mb + i;
        float d = SQQ[m] + sqc - 2.0f * acc[i];
        if (ncol == row0 + m) d = -3.0e38f;      // self at distance 0, always kept
        D2[m * 1024 + lcol] = d;
      }
    }
    __syncthreads();
    if (tid < 16) {
      for (int c = 0; c < 1024; ++c) {
        float v = D2[tid * 1024 + c];
        if (v < bestd[KNN - 1]) {
          int p = KNN - 1;
          while (p > 0 && bestd[p - 1] > v) {
            bestd[p] = bestd[p - 1]; besti[p] = besti[p - 1]; --p;
          }
          bestd[p] = v; besti[p] = ch * 1024 + c;
        }
      }
    }
    __syncthreads();
  }
  if (tid < 16) {
    for (int j = 0; j < KNN; ++j)
      idxout[(size_t)(b * N_ + row0 + tid) * KNN + j] = besti[j];
  }
}

// ---------------------------------------------------------------------------
// DynamicEdgeConv: NPW=4 nodes per workgroup (4 M-tiles of 16 edge rows).
// Each weight fragment is loaded once and used for 4 WMMAs -> 4x L2 intensity.
// layer1 512->512, layer2 512->256, max over 16 edge rows, + |g|^2 per node.
// Block = 256 threads (8 waves); grid = NODES/4. LDS ~136 KB.
// ---------------------------------------------------------------------------
__global__ void edgeconv_kernel(const half_t* __restrict__ X, const int* __restrict__ idx,
                                const half_t* __restrict__ W1fr, const float* __restrict__ b1,
                                const half_t* __restrict__ W2fr, const float* __restrict__ b2,
                                half_t* __restrict__ gout, float* __restrict__ sqout) {
  __shared__ half_t Ae[NPW * 16 * D4E];   // 64 KB edge features [x_i, x_j - x_i]
  __shared__ half_t Mid[NPW * 16 * D4E];  // 64 KB layer-1 activations (f16)
  __shared__ float  Red[NPW * 512];       // 8 KB
  __shared__ int    nbr[NPW * 16];
  int node0 = blockIdx.x * NPW;
  int b = node0 >> 11;                    // all NPW nodes share one batch (N_%NPW==0)
  int tid = threadIdx.x, lane = tid & 31, wave = tid >> 5;

  if (tid < NPW * 16) nbr[tid] = idx[(size_t)node0 * KNN + tid];
  __syncthreads();

  for (int e = tid; e < NPW * 16 * D4E; e += 256) {
    int p = e >> 13, rem = e & 8191;
    int j = rem >> 9, c = rem & 511, cc = c & 255;
    int node = node0 + p;
    float xv = (float)X[(size_t)node * D2E + cc];
    float val = (c < D2E) ? xv
              : (float)X[((size_t)b * N_ + nbr[p * 16 + j]) * D2E + cc] - xv;
    Ae[p * 8192 + j * D4E + c] = (half_t)val;
  }
  __syncthreads();

  // layer 1: [16x512] @ [512x512] (BN folded) + ReLU -> Mid, 4 M-tiles per B-frag
  for (int t = 0; t < 4; ++t) {
    int nt = wave * 4 + t;
    v8f acc[NPW];
#pragma unroll
    for (int p = 0; p < NPW; ++p) acc[p] = (v8f){0.f,0.f,0.f,0.f,0.f,0.f,0.f,0.f};
#pragma unroll
    for (int kt = 0; kt < 16; ++kt) {
      v16h w = load_bfrag(W1fr, 32, kt, nt, lane);
#pragma unroll
      for (int p = 0; p < NPW; ++p) {
        v16h a = load_afrag(Ae + p * 8192, D4E, lane, kt);
        acc[p] = wmma16(a, w, acc[p]);
      }
    }
    int col = nt * 16 + (lane & 15);
    float bb = b1[col];
    int mb = (lane & 16) ? 8 : 0;
#pragma unroll
    for (int p = 0; p < NPW; ++p)
#pragma unroll
      for (int i = 0; i < 8; ++i) {
        float v = acc[p][i] + bb;
        v = v > 0.f ? v : 0.f;
        Mid[p * 8192 + (mb + i) * D4E + col] = (half_t)v;
      }
  }
  __syncthreads();

  // layer 2: [16x512] @ [512x256] + ReLU, then max over the 16 edge rows
  for (int t = 0; t < 2; ++t) {
    int nt = wave * 2 + t;
    v8f acc[NPW];
#pragma unroll
    for (int p = 0; p < NPW; ++p) acc[p] = (v8f){0.f,0.f,0.f,0.f,0.f,0.f,0.f,0.f};
#pragma unroll
    for (int kt = 0; kt < 16; ++kt) {
      v16h w = load_bfrag(W2fr, 16, kt, nt, lane);
#pragma unroll
      for (int p = 0; p < NPW; ++p) {
        v16h a = load_afrag(Mid + p * 8192, D4E, lane, kt);
        acc[p] = wmma16(a, w, acc[p]);
      }
    }
    int col = nt * 16 + (lane & 15);
    float bb = b2[col];
    int hi = (lane & 16) ? 256 : 0;
#pragma unroll
    for (int p = 0; p < NPW; ++p) {
      float mx = -3.0e38f;
#pragma unroll
      for (int i = 0; i < 8; ++i) {
        float v = acc[p][i] + bb;
        v = v > 0.f ? v : 0.f;
        mx = v > mx ? v : mx;
      }
      Red[p * 512 + col + hi] = mx;            // rows 0-7 vs 8-15 halves
    }
  }
  __syncthreads();

  float v[NPW];
#pragma unroll
  for (int p = 0; p < NPW; ++p) {
    v[p] = fmaxf(Red[p * 512 + tid], Red[p * 512 + tid + 256]);
    gout[(size_t)(node0 + p) * D2E + tid] = (half_t)v[p];
  }
  __syncthreads();
#pragma unroll
  for (int p = 0; p < NPW; ++p) Red[p * 512 + tid] = v[p] * v[p];
  __syncthreads();
  for (int s = 128; s > 0; s >>= 1) {
    if (tid < s) {
#pragma unroll
      for (int p = 0; p < NPW; ++p) Red[p * 512 + tid] += Red[p * 512 + tid + s];
    }
    __syncthreads();
  }
  if (tid == 0) {
#pragma unroll
    for (int p = 0; p < NPW; ++p) sqout[node0 + p] = Red[p * 512];
  }
}

// ---------------------------------------------------------------------------
// feat MLP: [concat(g1,g2)] 512 -> 512 -> 256, ReLU both layers (no BN).
// NPW=4 node-tiles (64 rows) per workgroup; output f32 straight into d_out.
// ---------------------------------------------------------------------------
__global__ void featmlp_kernel(const half_t* __restrict__ g1, const half_t* __restrict__ g2,
                               const half_t* __restrict__ W1fr, const float* __restrict__ b1,
                               const half_t* __restrict__ W2fr, const float* __restrict__ b2,
                               float* __restrict__ fout) {
  __shared__ half_t Ain[NPW * 16 * D4E];  // 64 KB
  __shared__ half_t Mid[NPW * 16 * D4E];  // 64 KB
  int tile0 = blockIdx.x * (NPW * 16);
  int tid = threadIdx.x, lane = tid & 31, wave = tid >> 5;

  for (int e = tid; e < NPW * 16 * D4E; e += 256) {
    int p = e >> 13, rem = e & 8191;
    int r = rem >> 9, c = rem & 511;
    int row = tile0 + p * 16 + r;
    half_t v = (c < D2E) ? g1[(size_t)row * D2E + c]
                         : g2[(size_t)row * D2E + (c - D2E)];
    Ain[p * 8192 + r * D4E + c] = v;
  }
  __syncthreads();

  for (int t = 0; t < 4; ++t) {
    int nt = wave * 4 + t;
    v8f acc[NPW];
#pragma unroll
    for (int p = 0; p < NPW; ++p) acc[p] = (v8f){0.f,0.f,0.f,0.f,0.f,0.f,0.f,0.f};
#pragma unroll
    for (int kt = 0; kt < 16; ++kt) {
      v16h w = load_bfrag(W1fr, 32, kt, nt, lane);
#pragma unroll
      for (int p = 0; p < NPW; ++p) {
        v16h a = load_afrag(Ain + p * 8192, D4E, lane, kt);
        acc[p] = wmma16(a, w, acc[p]);
      }
    }
    int col = nt * 16 + (lane & 15);
    float bb = b1[col];
    int mb = (lane & 16) ? 8 : 0;
#pragma unroll
    for (int p = 0; p < NPW; ++p)
#pragma unroll
      for (int i = 0; i < 8; ++i) {
        float v = acc[p][i] + bb;
        v = v > 0.f ? v : 0.f;
        Mid[p * 8192 + (mb + i) * D4E + col] = (half_t)v;
      }
  }
  __syncthreads();

  for (int t = 0; t < 2; ++t) {
    int nt = wave * 2 + t;
    v8f acc[NPW];
#pragma unroll
    for (int p = 0; p < NPW; ++p) acc[p] = (v8f){0.f,0.f,0.f,0.f,0.f,0.f,0.f,0.f};
#pragma unroll
    for (int kt = 0; kt < 16; ++kt) {
      v16h w = load_bfrag(W2fr, 16, kt, nt, lane);
#pragma unroll
      for (int p = 0; p < NPW; ++p) {
        v16h a = load_afrag(Mid + p * 8192, D4E, lane, kt);
        acc[p] = wmma16(a, w, acc[p]);
      }
    }
    int col = nt * 16 + (lane & 15);
    float bb = b2[col];
    int mb = (lane & 16) ? 8 : 0;
#pragma unroll
    for (int p = 0; p < NPW; ++p)
#pragma unroll
      for (int i = 0; i < 8; ++i) {
        float v = acc[p][i] + bb;
        v = v > 0.f ? v : 0.f;
        fout[(size_t)(tile0 + p * 16 + mb + i) * D2E + col] = v;
      }
  }
}

// ---------------------------------------------------------------------------
// Small heads (GEMV-shaped): one wave per node, LDS staging. ReLU every layer.
// ---------------------------------------------------------------------------
__global__ void heads_kernel(const float* __restrict__ f,
    const float* Wr1, const float* br1, const float* Wr2, const float* br2,
    const float* Wc1, const float* bc1, const float* Wc2, const float* bc2,
    const float* Wc3, const float* bc3,
    const float* Wo1, const float* bo1, const float* Wo2, const float* bo2,
    const float* Wo3, const float* bo3,
    float* __restrict__ refout, float* __restrict__ clsout, float* __restrict__ offout) {
  __shared__ float fx[8 * 256];
  __shared__ float hb[8 * 128];
  __shared__ float h2b[8 * 64];
  __shared__ float part[8 * 32];
  int tid = threadIdx.x, lane = tid & 31, wave = tid >> 5;
  int node = blockIdx.x * 8 + wave;

  for (int j = 0; j < 8; ++j)
    fx[wave * 256 + lane * 8 + j] = f[(size_t)node * 256 + lane * 8 + j];
  __syncthreads();

  // ref: 256 -> 128 -> 1
  for (int o = lane; o < 128; o += 32) {
    float v = br1[o];
    for (int c = 0; c < 256; ++c) v += fx[wave * 256 + c] * Wr1[c * 128 + o];
    hb[wave * 128 + o] = v > 0.f ? v : 0.f;
  }
  __syncthreads();
  {
    float p = 0.f;
    for (int o = lane; o < 128; o += 32) p += hb[wave * 128 + o] * Wr2[o];
    part[wave * 32 + lane] = p;
  }
  __syncthreads();
  if (lane == 0) {
    float s = br2[0];
    for (int l = 0; l < 32; ++l) s += part[wave * 32 + l];
    refout[node] = s > 0.f ? s : 0.f;
  }
  __syncthreads();

  // objcls: 256 -> 128 -> 16 -> 21
  for (int o = lane; o < 128; o += 32) {
    float v = bc1[o];
    for (int c = 0; c < 256; ++c) v += fx[wave * 256 + c] * Wc1[c * 128 + o];
    hb[wave * 128 + o] = v > 0.f ? v : 0.f;
  }
  __syncthreads();
  if (lane < 16) {
    float v = bc2[lane];
    for (int o = 0; o < 128; ++o) v += hb[wave * 128 + o] * Wc2[o * 16 + lane];
    h2b[wave * 64 + lane] = v > 0.f ? v : 0.f;
  }
  __syncthreads();
  if (lane < C_) {
    float v = bc3[lane];
    for (int o = 0; o < 16; ++o) v += h2b[wave * 64 + o] * Wc3[o * C_ + lane];
    clsout[(size_t)node * C_ + lane] = v > 0.f ? v : 0.f;
  }
  __syncthreads();

  // offset: 256 -> 128 -> 64 -> 2
  for (int o = lane; o < 128; o += 32) {
    float v = bo1[o];
    for (int c = 0; c < 256; ++c) v += fx[wave * 256 + c] * Wo1[c * 128 + o];
    hb[wave * 128 + o] = v > 0.f ? v : 0.f;
  }
  __syncthreads();
  for (int o = lane; o < 64; o += 32) {
    float v = bo2[o];
    for (int oo = 0; oo < 128; ++oo) v += hb[wave * 128 + oo] * Wo2[oo * 64 + o];
    h2b[wave * 64 + o] = v > 0.f ? v : 0.f;
  }
  __syncthreads();
  if (lane < 2) {
    float v = bo3[lane];
    for (int o = 0; o < 64; ++o) v += h2b[wave * 64 + o] * Wo3[o * 2 + lane];
    offout[(size_t)node * 2 + lane] = v > 0.f ? v : 0.f;
  }
}

// tgt: desc [B,128] -> 16 -> 21, ReLU both layers. Tiny.
__global__ void tgt_kernel(const float* __restrict__ desc,
                           const float* W1, const float* b1,
                           const float* W2, const float* b2,
                           float* __restrict__ out) {
  int t = threadIdx.x;
  if (t >= B_) return;
  float h[16];
#pragma unroll
  for (int o = 0; o < 16; ++o) {
    float v = b1[o];
    for (int c = 0; c < 128; ++c) v += desc[t * 128 + c] * W1[c * 16 + o];
    h[o] = v > 0.f ? v : 0.f;
  }
  for (int j = 0; j < C_; ++j) {
    float v = b2[j];
#pragma unroll
    for (int o = 0; o < 16; ++o) v += h[o] * W2[o * C_ + j];
    out[t * C_ + j] = v > 0.f ? v : 0.f;
  }
}

// ---------------------------------------------------------------------------
extern "C" void kernel_launch(void* const* d_in, const int* in_sizes, int n_in,
                              void* d_out, int out_size, void* d_ws, size_t ws_size,
                              hipStream_t stream) {
  // --- inputs (setup_inputs dict order, leaf arrays flattened in order)
  const int*   ci   = (const int*)d_in[0];
  const float* pos  = (const float*)d_in[1];
  const float* desc = (const float*)d_in[2];
  const float* ctab = (const float*)d_in[3];
  const float* pW0 = (const float*)d_in[4],  *pb0 = (const float*)d_in[5];
  const float* pW1 = (const float*)d_in[6],  *pb1 = (const float*)d_in[7];
  // graph1: (W,b,g,beta) x2 -> 8..15 ; graph2 -> 16..23 ; feat -> 24..27
  // ref 28..31, tgt 32..35, objcls 36..41, off 42..47, k 48 (unused, fixed 16)

  // --- workspace carve (bump allocator, 256B aligned). ~29 MB total.
  char* ws = (char*)d_ws;
  size_t cur = 0;
  auto carve = [&](size_t bytes) -> void* {
    void* p = ws + cur;
    cur += (bytes + 255) & ~(size_t)255;
    return p;
  };
  half_t* feats = (half_t*)carve((size_t)NODES * D2E * 2);
  half_t* g1buf = (half_t*)carve((size_t)NODES * D2E * 2);
  half_t* g2buf = (half_t*)carve((size_t)NODES * D2E * 2);
  float*  sq0   = (float*)carve((size_t)NODES * 4);
  float*  sq1   = (float*)carve((size_t)NODES * 4);
  int*    idx   = (int*)carve((size_t)NODES * KNN * 4);
  half_t* g1W1fr = (half_t*)carve(512 * 512 * 2); float* g1b1f = (float*)carve(512 * 4);
  half_t* g1W2fr = (half_t*)carve(512 * 256 * 2); float* g1b2f = (float*)carve(256 * 4);
  half_t* g2W1fr = (half_t*)carve(512 * 512 * 2); float* g2b1f = (float*)carve(512 * 4);
  half_t* g2W2fr = (half_t*)carve(512 * 256 * 2); float* g2b2f = (float*)carve(256 * 4);
  half_t* fW1fr  = (half_t*)carve(512 * 512 * 2); float* fb1f  = (float*)carve(512 * 4);
  half_t* fW2fr  = (half_t*)carve(512 * 256 * 2); float* fb2f  = (float*)carve(256 * 4);

  // --- output layout (f, ref, tgt, cls, off) concatenated flat
  float* outf   = (float*)d_out;
  float* fout   = outf;                         // 8*2048*256
  float* refout = outf + (size_t)NODES * D2E;   // 8*2048
  float* tgtout = refout + NODES;               // 8*21
  float* clsout = tgtout + B_ * C_;             // 8*2048*21
  float* offout = clsout + (size_t)NODES * C_;  // 8*2048*2

  // --- 1) fold + swizzle weights
  dim3 pb(256);
  prep_weights<<<dim3((512 * 512 + 255) / 256), pb, 0, stream>>>(
      (const float*)d_in[8], (const float*)d_in[9], (const float*)d_in[10],
      (const float*)d_in[11], 512, 512, g1W1fr, g1b1f);
  prep_weights<<<dim3((512 * 256 + 255) / 256), pb, 0, stream>>>(
      (const float*)d_in[12], (const float*)d_in[13], (const float*)d_in[14],
      (const float*)d_in[15], 512, 256, g1W2fr, g1b2f);
  prep_weights<<<dim3((512 * 512 + 255) / 256), pb, 0, stream>>>(
      (const float*)d_in[16], (const float*)d_in[17], (const float*)d_in[18],
      (const float*)d_in[19], 512, 512, g2W1fr, g2b1f);
  prep_weights<<<dim3((512 * 256 + 255) / 256), pb, 0, stream>>>(
      (const float*)d_in[20], (const float*)d_in[21], (const float*)d_in[22],
      (const float*)d_in[23], 512, 256, g2W2fr, g2b2f);
  prep_weights<<<dim3((512 * 512 + 255) / 256), pb, 0, stream>>>(
      (const float*)d_in[24], (const float*)d_in[25], nullptr, nullptr,
      512, 512, fW1fr, fb1f);
  prep_weights<<<dim3((512 * 256 + 255) / 256), pb, 0, stream>>>(
      (const float*)d_in[26], (const float*)d_in[27], nullptr, nullptr,
      512, 256, fW2fr, fb2f);

  // --- 2) encode nodes
  encode_kernel<<<dim3((NODES + 255) / 256), dim3(256), 0, stream>>>(
      ci, pos, desc, ctab, pW0, pb0, pW1, pb1, feats, sq0);

  // --- 3) graph layer 1: kNN + edge conv
  knn_kernel<<<dim3(B_ * (N_ / 16)), dim3(512), 0, stream>>>(feats, sq0, idx);
  edgeconv_kernel<<<dim3(NODES / NPW), dim3(256), 0, stream>>>(
      feats, idx, g1W1fr, g1b1f, g1W2fr, g1b2f, g1buf, sq1);

  // --- 4) graph layer 2 (kNN recomputed in g1 feature space)
  knn_kernel<<<dim3(B_ * (N_ / 16)), dim3(512), 0, stream>>>(g1buf, sq1, idx);
  edgeconv_kernel<<<dim3(NODES / NPW), dim3(256), 0, stream>>>(
      g1buf, idx, g2W1fr, g2b1f, g2W2fr, g2b2f, g2buf, sq0);

  // --- 5) feature MLP over concat(g1,g2) -> f (f32, into d_out)
  featmlp_kernel<<<dim3(NODES / (NPW * 16)), dim3(256), 0, stream>>>(
      g1buf, g2buf, fW1fr, fb1f, fW2fr, fb2f, fout);

  // --- 6) heads
  heads_kernel<<<dim3(NODES / 8), dim3(256), 0, stream>>>(
      fout,
      (const float*)d_in[28], (const float*)d_in[29],
      (const float*)d_in[30], (const float*)d_in[31],
      (const float*)d_in[36], (const float*)d_in[37],
      (const float*)d_in[38], (const float*)d_in[39],
      (const float*)d_in[40], (const float*)d_in[41],
      (const float*)d_in[42], (const float*)d_in[43],
      (const float*)d_in[44], (const float*)d_in[45],
      (const float*)d_in[46], (const float*)d_in[47],
      refout, clsout, offout);
  tgt_kernel<<<dim3(1), dim3(32), 0, stream>>>(
      desc, (const float*)d_in[32], (const float*)d_in[33],
      (const float*)d_in[34], (const float*)d_in[35], tgtout);
}
